// NextPanelHead_33208687133254
// MI455X (gfx1250) — matrix operand order repaired and checked by reading, not compiled, not run on hardware
//
#include <hip/hip_runtime.h>
#include <hip/hip_bf16.h>

// ---------------------------------------------------------------------------
// CDNA5 / gfx1250 bf16 WMMA GEMM pipeline for logits = (P @ W) @ P^T + bias
//   B=8, N=2048, D=384   (all tile sizes divide evenly; no bounds checks)
// Async global->LDS tile copies (ASYNCcnt) + LDS double buffering.
// ---------------------------------------------------------------------------

typedef __attribute__((ext_vector_type(16))) __bf16        v16bf;
typedef __attribute__((ext_vector_type(8)))  float         v8f;
typedef __attribute__((ext_vector_type(4)))  unsigned int  u32x4;

// GCC-style vector type exactly matching the async-LDS builtin's pointee type.
typedef int gv4i __attribute__((vector_size(16)));

union FragCast { u32x4 u[2]; v16bf v; };

__device__ __forceinline__ __bf16 f2bf(float f) {
    // round-to-nearest-even f32 -> bf16
    unsigned u = __builtin_bit_cast(unsigned, f);
    unsigned r = (u + 0x7FFFu + ((u >> 16) & 1u)) >> 16;
    unsigned short h = (unsigned short)r;
    return __builtin_bit_cast(__bf16, h);
}

// ---- CDNA5 async global->LDS copy (16B per lane), ASYNCcnt-tracked --------
__device__ __forceinline__ void async_copy16(const void* gptr, void* lptr) {
#if __has_builtin(__builtin_amdgcn_global_load_async_to_lds_b128)
    __builtin_amdgcn_global_load_async_to_lds_b128(
        (__attribute__((address_space(1))) gv4i*)gptr,
        (__attribute__((address_space(3))) gv4i*)lptr, 0, 0);
#else
    unsigned lds = (unsigned)(unsigned long long)lptr;   // low 32 bits = LDS addr
    unsigned long long ga = (unsigned long long)gptr;
    asm volatile("global_load_async_to_lds_b128 %0, %1, off"
                 :: "v"(lds), "v"(ga) : "memory");
#endif
}

__device__ __forceinline__ void wait_asynccnt0() {
#if __has_builtin(__builtin_amdgcn_s_wait_asynccnt)
    __builtin_amdgcn_s_wait_asynccnt(0);
#else
    asm volatile("s_wait_asynccnt 0x0" ::: "memory");
#endif
}
__device__ __forceinline__ void wait_asynccnt4() {
#if __has_builtin(__builtin_amdgcn_s_wait_asynccnt)
    __builtin_amdgcn_s_wait_asynccnt(4);
#else
    asm volatile("s_wait_asynccnt 0x4" ::: "memory");
#endif
}

// -------------------- elementwise f32 -> bf16 ------------------------------
__global__ __launch_bounds__(256) void cvt_f32_bf16(const float* __restrict__ x,
                                                    __bf16* __restrict__ y, int n) {
    int i = blockIdx.x * 256 + threadIdx.x;
    if (i < n) y[i] = f2bf(x[i]);
}

// -------------------- W (DxD f32) -> W^T (DxD bf16) ------------------------
__global__ __launch_bounds__(256) void cvt_transpose_bf16(const float* __restrict__ W,
                                                          __bf16* __restrict__ Wt, int D) {
    int i = blockIdx.x * 256 + threadIdx.x;
    if (i < D * D) {
        int r = i / D, c = i % D;
        Wt[(long)c * D + r] = f2bf(W[(long)r * D + c]);
    }
}

// ---------------------------------------------------------------------------
// C[b] = A[b] (MxK, row-major bf16)  x  Bt[b]^T (Bt is NxK row-major bf16)
// Block tile 128x128, BK=32, double-buffered LDS. 256 threads = 8 wave32
// (2x4), wave tile 64x32 = 8 x v_wmma_f32_16x16x32_bf16 per K-step.
// ---------------------------------------------------------------------------
#define BM 128
#define BN 128
#define BK 32

template <bool OUT_BF16>
__global__ __launch_bounds__(256)
void gemm_nt_wmma_bf16(const __bf16* __restrict__ A,
                       const __bf16* __restrict__ Bt,
                       void* __restrict__ Cv,
                       const float* __restrict__ biasp,
                       int K, int ldc,
                       long strideA, long strideB, long strideC) {
    __shared__ __align__(16) __bf16 Asmem[2][BM * BK];   // [buf][row][k]  2x8 KB
    __shared__ __align__(16) __bf16 Bsmem[2][BN * BK];   // [buf][col][k]  2x8 KB

    const int tid  = threadIdx.x;
    const int lane = tid & 31;           // wave32
    const int wave = tid >> 5;           // 0..7
    const int wm   = wave >> 2;          // 0..1  (64 rows each)
    const int wn   = wave & 3;           // 0..3  (32 cols each)

    const int bm0   = blockIdx.y * BM;
    const int bn0   = blockIdx.x * BN;
    const int batch = blockIdx.z;

    const __bf16* Ab  = A  + (long)batch * strideA;
    const __bf16* Btb = Bt + (long)batch * strideB;

    const int krow = K >> 3;             // u32x4 (16B = 8 bf16) per global row
    const int NT   = K / BK;             // 12 K-tiles

    // Per-thread copy slots: 2 x 16B for A tile, 2 x 16B for B tile.
    // 4 async ops per wave-issue per tile pair.
    auto issue_tile = [&](int kt, int buf) {
        const u32x4* Ag  = (const u32x4*)Ab;
        const u32x4* Bg  = (const u32x4*)Btb;
        u32x4*       As4 = (u32x4*)Asmem[buf];
        u32x4*       Bs4 = (u32x4*)Bsmem[buf];
        const int kseg0 = (kt * BK) >> 3;
#pragma unroll
        for (int t = 0; t < 2; ++t) {
            int i   = tid + t * 256;     // 0..511
            int row = i >> 2;            // 0..127
            int seg = i & 3;             // 16B segment within 32 k's
            async_copy16(&Ag[(long)(bm0 + row) * krow + kseg0 + seg], &As4[i]);
            async_copy16(&Bg[(long)(bn0 + row) * krow + kseg0 + seg], &Bs4[i]);
        }
    };

    v8f acc[4][2] = {};                  // 64x32 per wave in 16x16 WMMA tiles

    issue_tile(0, 0);

    for (int kt = 0; kt < NT; ++kt) {
        const int buf = kt & 1;
        if (kt + 1 < NT) {
            issue_tile(kt + 1, buf ^ 1);
            wait_asynccnt4();            // 4 oldest (current tile) done, in order
        } else {
            wait_asynccnt0();
        }
        __syncthreads();                 // all waves' tile data visible in LDS

        // Per-wave fragments (layouts per CDNA5 ISA 7.12.2).
        const int rsel = lane & 15;
        const int half = lane >> 4;
        const __bf16* As = Asmem[buf];
        const __bf16* Bs = Bsmem[buf];
        v16bf afrag[4], bfrag[2];
#pragma unroll
        for (int mi = 0; mi < 4; ++mi) {
            // A 16x32 bf16: lane row = rsel, K = {half*8..+7, 16+half*8..+7}
            int row = wm * 64 + mi * 16 + rsel;
            FragCast f;
            f.u[0] = *(const u32x4*)&As[row * BK + half * 8];
            f.u[1] = *(const u32x4*)&As[row * BK + 16 + half * 8];
            afrag[mi] = f.v;
        }
#pragma unroll
        for (int ni = 0; ni < 2; ++ni) {
            // B 32x16 bf16: lane col = rsel, K = {half*8..+7, 16+half*8..+7}
            int col = wn * 32 + ni * 16 + rsel;
            FragCast f;
            f.u[0] = *(const u32x4*)&Bs[col * BK + half * 8];
            f.u[1] = *(const u32x4*)&Bs[col * BK + 16 + half * 8];
            bfrag[ni] = f.v;
        }

#pragma unroll
        for (int mi = 0; mi < 4; ++mi)
#pragma unroll
            for (int ni = 0; ni < 2; ++ni)
                acc[mi][ni] = __builtin_amdgcn_wmma_f32_16x16x32_bf16(
                    /*neg_a=*/false, afrag[mi],
                    /*neg_b=*/false, bfrag[ni],
                    /*c_mod=*/(short)0, acc[mi][ni],
                    /*reuse_a=*/false, /*reuse_b=*/false);

        __syncthreads();                 // done reading buf before it is re-filled
    }

    const float bias = OUT_BF16 ? 0.0f : biasp[0];

    // C/D layout: VGPR r, lanes 0-15 -> M = r, lanes 16-31 -> M = 8 + r.
#pragma unroll
    for (int mi = 0; mi < 4; ++mi) {
#pragma unroll
        for (int ni = 0; ni < 2; ++ni) {
            int row0 = bm0 + wm * 64 + mi * 16 + (lane >> 4) * 8;
            int col  = bn0 + wn * 32 + ni * 16 + (lane & 15);
            if (OUT_BF16) {
                __bf16* C = (__bf16*)Cv + (long)batch * strideC;
#pragma unroll
                for (int r = 0; r < 8; ++r)
                    C[(long)(row0 + r) * ldc + col] = f2bf(acc[mi][ni][r]);
            } else {
                float* C = (float*)Cv + (long)batch * strideC;
#pragma unroll
                for (int r = 0; r < 8; ++r)
                    C[(long)(row0 + r) * ldc + col] = acc[mi][ni][r] + bias;
            }
        }
    }
}

// ---------------------------------------------------------------------------
extern "C" void kernel_launch(void* const* d_in, const int* in_sizes, int n_in,
                              void* d_out, int out_size, void* d_ws, size_t ws_size,
                              hipStream_t stream) {
    (void)in_sizes; (void)n_in; (void)out_size; (void)ws_size;

    constexpr int Bsz = 8, N = 2048, D = 384;

    const float* P    = (const float*)d_in[0];   // [B, N, D]
    const float* W    = (const float*)d_in[1];   // [D, D]
    const float* bias = (const float*)d_in[2];   // [1]
    float*       out  = (float*)d_out;           // [B, N, N]

    // Workspace layout (bf16): P_bf16 | W^T_bf16 | PW_bf16   (~24.3 MB total)
    __bf16* Pb  = (__bf16*)d_ws;
    __bf16* Wt  = Pb + (long)Bsz * N * D;
    __bf16* PWb = Wt + (long)D * D;

    // 1) precision conversion
    cvt_f32_bf16<<<(Bsz * N * D + 255) / 256, 256, 0, stream>>>(P, Pb, Bsz * N * D);
    cvt_transpose_bf16<<<(D * D + 255) / 256, 256, 0, stream>>>(W, Wt, D);

    // 2) PW[b] = P[b] @ W          (M=2048, Nn=384, K=384) -> bf16
    {
        dim3 grid(D / BN, N / BM, Bsz);
        gemm_nt_wmma_bf16<true><<<grid, 256, 0, stream>>>(
            Pb, Wt, (void*)PWb, nullptr,
            /*K=*/D, /*ldc=*/D,
            /*strideA=*/(long)N * D, /*strideB=*/0L, /*strideC=*/(long)N * D);
    }

    // 3) out[b] = PW[b] @ P[b]^T + bias   (M=2048, Nn=2048, K=384) -> f32
    {
        dim3 grid(N / BN, N / BM, Bsz);
        gemm_nt_wmma_bf16<false><<<grid, 256, 0, stream>>>(
            PWb, Pb, (void*)out, bias,
            /*K=*/D, /*ldc=*/N,
            /*strideA=*/(long)N * D, /*strideB=*/(long)N * D, /*strideC=*/(long)N * N);
    }
}
